// SANet_58093727645753
// MI455X (gfx1250) — matrix operand order repaired and checked by reading, not compiled, not run on hardware
//
#include <hip/hip_runtime.h>
#include <hip/hip_bf16.h>
#include <stdint.h>

#define HW  4096
#define NC  512
#define NCH 256
#define NB  4

typedef __attribute__((ext_vector_type(16))) __bf16 v16bf;
typedef __attribute__((ext_vector_type(8)))  __bf16 v8bf;
typedef __attribute__((ext_vector_type(8)))  float  v8f;
typedef __attribute__((ext_vector_type(4)))  unsigned int v4u;
typedef __attribute__((ext_vector_type(8)))  int v8i;
typedef __attribute__((ext_vector_type(4)))  int v4i;

union frag_u { v16bf v; v8bf h[2]; };

// A-fragment (16x32 bf16): lane row = M, lanes 0-15 K0=0, lanes 16-31 K0=8,
// second 8-elem chunk at +16 elements.  row_base = start of the 32-wide K window.
__device__ __forceinline__ v16bf a_frag(const __bf16* row_base, int half) {
  frag_u u;
  const __bf16* p = row_base + half * 8;
  u.h[0] = *(const v8bf*)p;
  u.h[1] = *(const v8bf*)(p + 16);
  return u.v;
}
// B-fragment (32x16 bf16): lane column = N, lanes 0-15 K0=0, lanes 16-31 K0=8,
// second 8-elem chunk at +16 elements (same K order as A).
__device__ __forceinline__ v16bf b_frag(const __bf16* row_base, int half) {
  frag_u u;
  const __bf16* p = row_base + half * 8;
  u.h[0] = *(const v8bf*)p;
  u.h[1] = *(const v8bf*)(p + 16);
  return u.v;
}
__device__ __forceinline__ v8f wmma_bf16(v16bf a, v16bf b, v8f c) {
  return __builtin_amdgcn_wmma_f32_16x16x32_bf16(false, a, false, b, (short)0, c,
                                                 false, false);
}

// ---- async global->LDS (ASYNCcnt path) --------------------------------------
__device__ __forceinline__ void async_b128(uint32_t ldsoff, const void* gptr) {
  asm volatile("global_load_async_to_lds_b128 %0, %1, off"
               :: "v"(ldsoff), "v"((unsigned long long)(uintptr_t)gptr)
               : "memory");
}
__device__ __forceinline__ void wait_async_0() {
  asm volatile("s_wait_asynccnt 0x0" ::: "memory");
}

// ---- Tensor Data Mover: one-shot 2D tile -> LDS with hardware row padding ---
// pad_interval code: 0=2,1=4,2=8,3=16,...  dwords; pad_amount code: n -> n+1 dwords
__device__ __forceinline__ void tdm_load_2d(uint32_t lds_off, const void* gaddr,
                                            uint32_t tile_x, uint32_t tile_y,
                                            uint32_t tensor_x, uint32_t tensor_y,
                                            uint32_t stride_x,
                                            uint32_t pad_interval, uint32_t pad_amount)
{
  uint64_t ga = (uint64_t)(uintptr_t)gaddr;
  v4u g0;
  g0.x = 1u;                                   // count=1 (valid user descriptor)
  g0.y = lds_off;                              // lds_addr [63:32]
  g0.z = (uint32_t)ga;                         // global_addr lo
  g0.w = (uint32_t)(ga >> 32) | (2u << 30);    // global_addr hi | type=2 (image)
  v8i g1;
  g1[0] = (int)((1u << 16)                     // data_size = 2 bytes
              | (1u << 20)                     // pad_enable
              | (pad_interval << 22)
              | (pad_amount << 25));
  g1[1] = (int)((tensor_x & 0xffffu) << 16);   // tensor_dim0[15:0]
  g1[2] = (int)((tensor_x >> 16) | ((tensor_y & 0xffffu) << 16));
  g1[3] = (int)((tensor_y >> 16) | (tile_x << 16));    // tile_dim0
  g1[4] = (int)(tile_y & 0xffffu);             // tile_dim1 (tile_dim2 = 0)
  g1[5] = (int)stride_x;                       // tensor_dim0_stride lo32
  g1[6] = 0;
  g1[7] = 0;
  v4i z4 = {0, 0, 0, 0};
  v8i z8 = {0, 0, 0, 0, 0, 0, 0, 0};
  __builtin_amdgcn_tensor_load_to_lds(g0, g1, z4, z4, z8, 0);
}
__device__ __forceinline__ void wait_tensor_le2() {
  __builtin_amdgcn_s_wait_tensorcnt(2);
}
__device__ __forceinline__ void wait_tensor_0() {
  __builtin_amdgcn_s_wait_tensorcnt(0);
}

// =============================================================================
// Projection GEMM (1x1 conv), bf16 WMMA, fp32 accum.
// mode 0: D[i][o] = (X^T W^T + b) * outScale   (Ff, Fg: Odim=256)
// mode 1: D[o][i] =  W X + b                   (Fh:    Odim=512, c-major out)
// X: [NB][NC][HW] fp32 (c-major), W: [Odim][NC] fp32.
// =============================================================================
__global__ void __launch_bounds__(128)
proj_kernel(const float* __restrict__ X, const float* __restrict__ W,
            const float* __restrict__ bias, __bf16* __restrict__ D,
            int Odim, int mode, float outScale)
{
  extern __shared__ char smem[];
  __bf16* XL = (__bf16*)smem;            // [64 i][40] (32 c + pad)
  __bf16* WL = (__bf16*)(smem + 5120);   // [64 o][40]
  const int tid  = threadIdx.x;
  const int wave = tid >> 5, lane = tid & 31, half = lane >> 4, l16 = lane & 15;
  const int batch = blockIdx.z;
  int iBase, oBase;
  if (mode == 0) { iBase = blockIdx.x * 64; oBase = blockIdx.y * 64; }
  else           { oBase = blockIdx.x * 64; iBase = blockIdx.y * 64; }
  const float* Xb = X + (size_t)batch * NC * HW;
  const int mSub = (wave & 1) * 32, nSub = (wave >> 1) * 32;
  v8f acc[2][2] = {};

  for (int kk = 0; kk < NC / 32; ++kk) {
    __syncthreads();
    for (int q = tid; q < 2048; q += 128) {          // X tile -> XL[i][c] (bf16)
      int ci = q >> 6, ii = q & 63;
      XL[ii * 40 + ci] = (__bf16)Xb[(size_t)(kk * 32 + ci) * HW + iBase + ii];
    }
    for (int q = tid; q < 2048; q += 128) {          // W tile -> WL[o][c] (bf16)
      int oo = q >> 5, cc = q & 31;
      WL[oo * 40 + cc] = (__bf16)W[(size_t)(oBase + oo) * NC + kk * 32 + cc];
    }
    __syncthreads();
    const __bf16* AL = (mode == 0) ? XL : WL;
    const __bf16* BL = (mode == 0) ? WL : XL;
    v16bf a0 = a_frag(AL + (mSub + l16) * 40, half);
    v16bf a1 = a_frag(AL + (mSub + 16 + l16) * 40, half);
    v16bf b0 = b_frag(BL + (nSub + l16) * 40, half);
    v16bf b1 = b_frag(BL + (nSub + 16 + l16) * 40, half);
    acc[0][0] = wmma_bf16(a0, b0, acc[0][0]);
    acc[0][1] = wmma_bf16(a0, b1, acc[0][1]);
    acc[1][0] = wmma_bf16(a1, b0, acc[1][0]);
    acc[1][1] = wmma_bf16(a1, b1, acc[1][1]);
  }
  if (mode == 0) {
#pragma unroll
    for (int mi = 0; mi < 2; ++mi)
#pragma unroll
      for (int ni = 0; ni < 2; ++ni) {
        v8f r = acc[mi][ni];
        int o = oBase + nSub + ni * 16 + l16;
        float bo = bias[o];
        size_t base = ((size_t)batch * HW + iBase + mSub + mi * 16 + 8 * half) *
                          (size_t)Odim + o;
#pragma unroll
        for (int v = 0; v < 8; ++v)
          D[base + (size_t)v * Odim] = (__bf16)((r[v] + bo) * outScale);
      }
  } else {
#pragma unroll
    for (int mi = 0; mi < 2; ++mi)
#pragma unroll
      for (int ni = 0; ni < 2; ++ni) {
        v8f r = acc[mi][ni];
        int ob = oBase + mSub + mi * 16 + 8 * half;
        int i  = iBase + nSub + ni * 16 + l16;
#pragma unroll
        for (int v = 0; v < 8; ++v)
          D[((size_t)batch * Odim + ob + v) * (size_t)HW + i] =
              (__bf16)(r[v] + bias[ob + v]);
      }
  }
}

// =============================================================================
// Fused flash attention: O[b][i][c] = softmax_j(Ff[i]·Fg[j]) · Fh[c][j]
// WG = 256 thr (8 waves): wave = (rg = i-rowgroup 0..3, ch = c-half 0..1).
// LDS: Ff[64][264] | Fg x2 [32][264] | Fh x2 [512][40] | P [8][16][40]
// Ff staged once via ASYNCcnt path; Fg/Fh per-tile via TDM (TENSORcnt),
// double-buffered, hardware row padding matches the LDS layout.
// =============================================================================
#define L_FF 0
#define L_FG 33792
#define L_FH 67584
#define L_P  149504
#define ATTN_LDS 159744

__device__ __forceinline__ void issue_tile(const __bf16* FgG, const __bf16* FhG,
                                           int buf, int jt) {
  // Fg tile: 32 j-rows x 256 c, row 512B -> interval 128 dw (code 6), pad 4 dw (code 3)
  tdm_load_2d((uint32_t)L_FG + (uint32_t)buf * 16896u, FgG + (size_t)jt * 32 * NCH,
              /*tile_x=*/256, /*tile_y=*/32, /*tensor_x=*/256, /*tensor_y=*/HW,
              /*stride_x=*/256, /*interval=*/6, /*amount=*/3);
  // Fh tile: 512 c-rows x 32 j, row 64B -> interval 16 dw (code 3), pad 4 dw (code 3)
  tdm_load_2d((uint32_t)L_FH + (uint32_t)buf * 40960u, FhG + (size_t)jt * 32,
              /*tile_x=*/32, /*tile_y=*/512, /*tensor_x=*/HW, /*tensor_y=*/NC,
              /*stride_x=*/HW, /*interval=*/3, /*amount=*/3);
}

__global__ void __launch_bounds__(256)
attn_kernel(const __bf16* __restrict__ Ff, const __bf16* __restrict__ Fg,
            const __bf16* __restrict__ Fh, __bf16* __restrict__ O)
{
  extern __shared__ char smem[];
  __bf16* FfL = (__bf16*)(smem + L_FF);
  const int tid  = threadIdx.x;
  const int wave = tid >> 5, lane = tid & 31, half = lane >> 4, l16 = lane & 15;
  const int rg = wave & 3;    // i rows rg*16 .. rg*16+15
  const int ch = wave >> 2;   // c half: ch*256 ..
  const int batch = blockIdx.y;
  const int i0 = blockIdx.x * 64;

  const __bf16* FfG = Ff + ((size_t)batch * HW + i0) * NCH;
  const __bf16* FgG = Fg + (size_t)batch * HW * NCH;
  const __bf16* FhG = Fh + (size_t)batch * NC * HW;

  // prologue: stage this block's Ff rows once (ASYNC path), issue TDM tile 0
  for (int q = tid; q < 2048; q += 256) {
    int r = q >> 5, cc = q & 31;
    async_b128((uint32_t)(L_FF + r * 528 + cc * 16), FfG + (size_t)r * NCH + cc * 8);
  }
  if (wave == 0) issue_tile(FgG, FhG, 0, 0);
  wait_async_0();

  v8f Of[16] = {};
  float mrow[8], lrow[8];
#pragma unroll
  for (int v = 0; v < 8; ++v) { mrow[v] = -__builtin_inff(); lrow[v] = 0.f; }

  for (int t = 0; t < HW / 32; ++t) {
    __syncthreads();                       // all waves done reading buf t-1
    if (wave == 0) {
      if (t + 1 < HW / 32) {
        issue_tile(FgG, FhG, (t + 1) & 1, t + 1);
        wait_tensor_le2();                 // tile t landed (TDM in-order per wave)
      } else {
        wait_tensor_0();
      }
    }
    __syncthreads();                       // tile t visible to all waves
    const int buf = t & 1;
    __bf16* FgL = (__bf16*)(smem + L_FG + buf * 16896);
    __bf16* FhL = (__bf16*)(smem + L_FH + buf * 40960);
    __bf16* PL  = (__bf16*)(smem + L_P  + wave * 1280);

    // S(16x32) = Ff(rows rg*16..)(16x256) x Fg^T(256x32), K-steps of 32
    v8f S0 = {}, S1 = {};
#pragma unroll
    for (int k = 0; k < 8; ++k) {
      v16bf a  = a_frag(FfL + (rg * 16 + l16) * 264 + k * 32, half);
      v16bf b0 = b_frag(FgL + l16 * 264 + k * 32, half);
      v16bf b1 = b_frag(FgL + (16 + l16) * 264 + k * 32, half);
      S0 = wmma_bf16(a, b0, S0);
      S1 = wmma_bf16(a, b1, S1);
    }
    // online softmax over the 32 j of this tile (row = v + 8*half)
    float p0[8], p1[8], scl[8];
#pragma unroll
    for (int v = 0; v < 8; ++v) {
      float c = fmaxf(S0[v], S1[v]);
      c = fmaxf(c, __shfl_xor(c, 1));
      c = fmaxf(c, __shfl_xor(c, 2));
      c = fmaxf(c, __shfl_xor(c, 4));
      c = fmaxf(c, __shfl_xor(c, 8));
      float mn = fmaxf(mrow[v], c);
      scl[v] = __expf(mrow[v] - mn);
      p0[v]  = __expf(S0[v] - mn);
      p1[v]  = __expf(S1[v] - mn);
      float rs = p0[v] + p1[v];
      rs += __shfl_xor(rs, 1);
      rs += __shfl_xor(rs, 2);
      rs += __shfl_xor(rs, 4);
      rs += __shfl_xor(rs, 8);
      lrow[v] = lrow[v] * scl[v] + rs;
      mrow[v] = mn;
    }
#pragma unroll
    for (int ct = 0; ct < 16; ++ct)
#pragma unroll
      for (int v = 0; v < 8; ++v) Of[ct][v] *= scl[v];
    // C/D layout -> A layout via LDS bounce (per-wave private region)
#pragma unroll
    for (int v = 0; v < 8; ++v) {
      int r = v + 8 * half;
      PL[r * 40 + l16]      = (__bf16)p0[v];
      PL[r * 40 + 16 + l16] = (__bf16)p1[v];
    }
    v16bf pA = a_frag(PL + l16 * 40, half);
    // O(16 x 256) += P(16x32) x Fh^T(32 x c), c-tile = ch*256 + ct*16
#pragma unroll
    for (int ct = 0; ct < 16; ++ct) {
      v16bf bh = b_frag(FhL + (ch * 256 + ct * 16 + l16) * 40, half);
      Of[ct] = wmma_bf16(pA, bh, Of[ct]);
    }
  }
  // normalize and store O as bf16 [b][i][c]
#pragma unroll
  for (int v = 0; v < 8; ++v) lrow[v] = 1.0f / lrow[v];
#pragma unroll
  for (int ct = 0; ct < 16; ++ct)
#pragma unroll
    for (int v = 0; v < 8; ++v) {
      int i = i0 + rg * 16 + v + 8 * half;
      int c = ch * 256 + ct * 16 + l16;
      O[((size_t)batch * HW + i) * NC + c] = (__bf16)(Of[ct][v] * lrow[v]);
    }
}

// =============================================================================
// Final 1x1 conv + residual: out[b][o][i] = out_w[o,:]·O[b][i][:] + b[o] + Fc
// A = O bf16 row-major (direct global b128 frags), B = out_w staged via LDS.
// Epilogue vectorized: accumulator index v maps to consecutive i.
// =============================================================================
__global__ void __launch_bounds__(128)
final_kernel(const __bf16* __restrict__ O, const float* __restrict__ Wout,
             const float* __restrict__ bout, const float* __restrict__ Fc,
             float* __restrict__ out)
{
  extern __shared__ char smem[];
  __bf16* WL = (__bf16*)smem;   // [64 o][40]
  const int tid  = threadIdx.x;
  const int wave = tid >> 5, lane = tid & 31, half = lane >> 4, l16 = lane & 15;
  const int batch = blockIdx.z;
  const int i0 = blockIdx.x * 64, o0 = blockIdx.y * 64;
  const __bf16* Ob = O + ((size_t)batch * HW + i0) * NC;
  const int mSub = (wave & 1) * 32, nSub = (wave >> 1) * 32;
  v8f acc[2][2] = {};

  for (int kk = 0; kk < NC / 32; ++kk) {
    __syncthreads();
    for (int q = tid; q < 2048; q += 128) {
      int oo = q >> 5, cc = q & 31;
      WL[oo * 40 + cc] = (__bf16)Wout[(size_t)(o0 + oo) * NC + kk * 32 + cc];
    }
    __syncthreads();
    v16bf a0 = a_frag(Ob + (size_t)(mSub + l16) * NC + kk * 32, half);
    v16bf a1 = a_frag(Ob + (size_t)(mSub + 16 + l16) * NC + kk * 32, half);
    v16bf b0 = b_frag(WL + (nSub + l16) * 40, half);
    v16bf b1 = b_frag(WL + (nSub + 16 + l16) * 40, half);
    acc[0][0] = wmma_bf16(a0, b0, acc[0][0]);
    acc[0][1] = wmma_bf16(a0, b1, acc[0][1]);
    acc[1][0] = wmma_bf16(a1, b0, acc[1][0]);
    acc[1][1] = wmma_bf16(a1, b1, acc[1][1]);
  }
#pragma unroll
  for (int mi = 0; mi < 2; ++mi)
#pragma unroll
    for (int ni = 0; ni < 2; ++ni) {
      v8f r = acc[mi][ni];
      int ib = i0 + mSub + mi * 16 + 8 * half;          // 8 consecutive i
      int o  = o0 + nSub + ni * 16 + l16;
      size_t idx = ((size_t)batch * NC + o) * HW + ib;
      float bo = bout[o];
      float4 f0 = *(const float4*)&Fc[idx];
      float4 f1 = *(const float4*)&Fc[idx + 4];
      float4 r0 = make_float4(r[0] + bo + f0.x, r[1] + bo + f0.y,
                              r[2] + bo + f0.z, r[3] + bo + f0.w);
      float4 r1 = make_float4(r[4] + bo + f1.x, r[5] + bo + f1.y,
                              r[6] + bo + f1.z, r[7] + bo + f1.w);
      *(float4*)&out[idx]     = r0;
      *(float4*)&out[idx + 4] = r1;
    }
}

extern "C" void kernel_launch(void* const* d_in, const int* in_sizes, int n_in,
                              void* d_out, int out_size, void* d_ws, size_t ws_size,
                              hipStream_t stream) {
  const float* Fc  = (const float*)d_in[0];
  const float* Fs  = (const float*)d_in[1];
  const float* f_w = (const float*)d_in[2];
  const float* f_b = (const float*)d_in[3];
  const float* g_w = (const float*)d_in[4];
  const float* g_b = (const float*)d_in[5];
  const float* h_w = (const float*)d_in[6];
  const float* h_b = (const float*)d_in[7];
  const float* o_w = (const float*)d_in[8];
  const float* o_b = (const float*)d_in[9];
  float* out = (float*)d_out;
  char* ws = (char*)d_ws;
  __bf16* wsFf = (__bf16*)(ws);                       // [B][HW][256]  8 MB
  __bf16* wsFg = (__bf16*)(ws + ((size_t)8 << 20));   // [B][HW][256]  8 MB
  __bf16* wsFh = (__bf16*)(ws + ((size_t)16 << 20));  // [B][512][HW] 16 MB
  __bf16* wsO  = (__bf16*)(ws + ((size_t)32 << 20));  // [B][HW][512] 16 MB

  dim3 blk(128);
  // scale 1/sqrt(d_k) = 1/16 folded into Ff
  proj_kernel<<<dim3(64, 4, NB), blk, 10240, stream>>>(Fc, f_w, f_b, wsFf, 256, 0, 0.0625f);
  proj_kernel<<<dim3(64, 4, NB), blk, 10240, stream>>>(Fs, g_w, g_b, wsFg, 256, 0, 1.0f);
  proj_kernel<<<dim3(8, 64, NB), blk, 10240, stream>>>(Fs, h_w, h_b, wsFh, 512, 1, 1.0f);

  (void)hipFuncSetAttribute(reinterpret_cast<const void*>(attn_kernel),
                            hipFuncAttributeMaxDynamicSharedMemorySize, ATTN_LDS);
  attn_kernel<<<dim3(64, NB), dim3(256), ATTN_LDS, stream>>>(wsFf, wsFg, wsFh, wsO);

  final_kernel<<<dim3(64, 8, NB), blk, 5120, stream>>>(wsO, o_w, o_b, Fc, out);
}
